// PLSTMCell_14937896255657
// MI455X (gfx1250) — matrix operand order repaired and checked by reading, not compiled, not run on hardware
//
#include <hip/hip_runtime.h>
#include <hip/hip_bf16.h>
#include <stdint.h>

// ---------------------------------------------------------------------------
// Problem constants (match reference)
// ---------------------------------------------------------------------------
constexpr int Bb = 64;            // batch
constexpr int Tt = 1024;          // time steps
constexpr int Dd = 768;           // input features
constexpr int Nn = 768;           // hidden size
constexpr int BT = Bb * Tt;       // 65536 rows for the input projection
constexpr int KT32 = Dd / 32;     // 24 k-tiles of 32 (bf16 WMMA K)
constexpr int NT16 = Nn / 16;     // 48 n-tiles of 16 per gate
constexpr int NTFLAT = 4 * NT16;  // 192 flat n-tiles (4 gates concatenated)

// WMMA vector types (CDNA5 wave32)
typedef __bf16 v16bf __attribute__((ext_vector_type(16)));
typedef float  v8f   __attribute__((ext_vector_type(8)));

// ---------------------------------------------------------------------------
// Workspace layout (bytes)
//   xb  : bf16(x)                 [BT][D]                 100,663,296
//   wkp : packed bf16 Wk          [4][NT16][KT32][512]      4,718,592
//   wrp : packed bf16 Wr          [4][NT16][KT32][512]      4,718,592
//   hb  : bf16 h double buffer    [2][B][N]                   196,608
//   c   : f32 cell state          [B][N]                      196,608
//   xg  : f32 gate preactivations [4][BT][N]              805,306,368
// ---------------------------------------------------------------------------
constexpr size_t OFF_XB  = 0;
constexpr size_t SZ_XB   = (size_t)BT * Dd * 2;
constexpr size_t OFF_WKP = OFF_XB + SZ_XB;
constexpr size_t SZ_WP   = (size_t)4 * Dd * Nn * 2;
constexpr size_t OFF_WRP = OFF_WKP + SZ_WP;
constexpr size_t OFF_HB  = OFF_WRP + SZ_WP;
constexpr size_t SZ_HB   = (size_t)2 * Bb * Nn * 2;
constexpr size_t OFF_C   = OFF_HB + SZ_HB;
constexpr size_t SZ_C    = (size_t)Bb * Nn * 4;
constexpr size_t OFF_XG  = OFF_C + SZ_C;

__device__ __forceinline__ float hsig(float x) {
    return fminf(fmaxf(0.2f * x + 0.5f, 0.0f), 1.0f);
}

// ---------------------------------------------------------------------------
// CDNA5 async copy: global -> LDS, 16B per lane, tracked by ASYNCcnt.
// lds_off is the low-32-bit LDS byte address (flat LDS aperture: addr[31:0]).
// ---------------------------------------------------------------------------
__device__ __forceinline__ void async_b128(uint32_t lds_off, const void* gaddr) {
    asm volatile("global_load_async_to_lds_b128 %0, %1, off"
                 :: "v"(lds_off), "v"(gaddr) : "memory");
}

__device__ __forceinline__ void wait_async0() {
#if __has_builtin(__builtin_amdgcn_s_wait_asynccnt)
    __builtin_amdgcn_s_wait_asynccnt(0);
#else
    asm volatile("s_wait_asynccnt 0x0" ::: "memory");
#endif
}

// ---------------------------------------------------------------------------
// f32 -> bf16 elementwise convert
// ---------------------------------------------------------------------------
__global__ void cvt_bf16_kernel(const float* __restrict__ in,
                                __bf16* __restrict__ out, int n) {
    int i = blockIdx.x * blockDim.x + threadIdx.x;
    if (i < n) out[i] = (__bf16)in[i];
}

// ---------------------------------------------------------------------------
// Pack a row-major [768 x 768] f32 weight matrix into the per-lane B-fragment
// layout for v_wmma_f32_16x16x32_bf16:
//   block(nt,kt) of 512 bf16; element = lane*16 + j holds
//   W[k = kt*32 + 8*(lane>=16) + (j<8 ? j : 16 + (j-8))][n = nt*16 + (lane&15)]
// ---------------------------------------------------------------------------
__global__ void pack_w_kernel(const float* __restrict__ W,
                              __bf16* __restrict__ out) {
    int idx = blockIdx.x * blockDim.x + threadIdx.x;
    const int total = NT16 * KT32 * 512;       // 589,824 per gate matrix
    if (idx >= total) return;
    int j    = idx & 15;
    int lane = (idx >> 4) & 31;
    int kt   = (idx >> 9) % KT32;
    int nt   = idx / (512 * KT32);
    int k = kt * 32 + ((lane >> 4) ? 8 : 0) + ((j < 8) ? j : 16 + (j - 8));
    int n = nt * 16 + (lane & 15);
    out[idx] = (__bf16)W[(size_t)k * Nn + n];
}

// ---------------------------------------------------------------------------
// Initialize recurrent state from h0/c0
// ---------------------------------------------------------------------------
__global__ void init_state_kernel(const float* __restrict__ h0,
                                  const float* __restrict__ c0,
                                  __bf16* __restrict__ hb0,
                                  float* __restrict__ c) {
    int i = blockIdx.x * blockDim.x + threadIdx.x;
    if (i < Bb * Nn) {
        hb0[i] = (__bf16)h0[i];
        c[i]   = c0[i];
    }
}

// ---------------------------------------------------------------------------
// Fragment loaders (ISA 7.12.2 layouts, wave32)
// ---------------------------------------------------------------------------
__device__ __forceinline__ v16bf load_a_frag(const __bf16* __restrict__ A,
                                             int m0, int lda, int kbase,
                                             int lane) {
    int m    = lane & 15;
    int koff = (lane >> 4) ? 8 : 0;
    const __bf16* ap = A + (size_t)(m0 + m) * lda + kbase + koff;
    v16bf a;
#pragma unroll
    for (int j = 0; j < 8; ++j) {
        a[j]     = ap[j];
        a[j + 8] = ap[16 + j];
    }
    return a;
}

__device__ __forceinline__ v16bf load_b_frag(const __bf16* __restrict__ Bp,
                                             int lane) {
    return *(const v16bf*)(Bp + (size_t)lane * 16);
}

// ---------------------------------------------------------------------------
// Input projection as one flat GEMM: [BT x 768] * [768 x 3072], bf16 WMMA,
// f32 accumulate.  Block = 256 threads (8 waves), block tile M=64 x Nflat=128.
// Per k-step both operand tiles are staged to LDS with async global->LDS
// copies, double-buffered so the copy of tile kt+1 overlaps the WMMAs of kt.
//   waves: wm = wave&3 (M subtile), wn = wave>>2 (N half, 4 n-tiles each)
// ---------------------------------------------------------------------------
__global__ void proj_gemm_kernel(const __bf16* __restrict__ xb,
                                 const __bf16* __restrict__ wkp,
                                 float* __restrict__ xg) {
    const int lane = threadIdx.x & 31;
    const int wave = threadIdx.x >> 5;
    const int wm   = wave & 3;
    const int wn   = wave >> 2;
    const int m0       = blockIdx.x * 64;
    const int ntflat0  = blockIdx.y * 8;

    // Fragment-order staging buffers: [buf][tile][lane][16 bf16]
    __shared__ __align__(32) __bf16 sA[2][4][32][16];   //  8 KB
    __shared__ __align__(32) __bf16 sB[2][8][32][16];   // 16 KB

    auto stage = [&](int buf, int kt) {
        const int t = threadIdx.x;
        // A: 256 chunks of 16B (mtile, lane_c, half)
        {
            int mtile  = t >> 6;
            int lane_c = (t >> 1) & 31;
            int half   = t & 1;
            int row = m0 + mtile * 16 + (lane_c & 15);
            int col = kt * 32 + ((lane_c >> 4) ? 8 : 0) + half * 16;
            const __bf16* g = xb + (size_t)row * Dd + col;
            uint32_t l = (uint32_t)(uintptr_t)&sA[buf][mtile][lane_c][half * 8];
            async_b128(l, g);
        }
        // B: 256 chunks of 32B (ntile, lane_c) straight from packed layout
        {
            int ntile  = t >> 5;
            int lane_c = t & 31;
            const __bf16* g =
                wkp + ((size_t)(ntflat0 + ntile) * KT32 + kt) * 512 + lane_c * 16;
            uint32_t l = (uint32_t)(uintptr_t)&sB[buf][ntile][lane_c][0];
            async_b128(l, g);
            async_b128(l + 16, g + 8);
        }
    };

    v8f acc[4];
#pragma unroll
    for (int i = 0; i < 4; ++i)
#pragma unroll
        for (int r = 0; r < 8; ++r) acc[i][r] = 0.0f;

    stage(0, 0);
    wait_async0();
    __syncthreads();

    for (int kt = 0; kt < KT32; ++kt) {
        const int buf = kt & 1;
        if (kt + 1 < KT32) stage(buf ^ 1, kt + 1);

        v16bf a = *(const v16bf*)&sA[buf][wm][lane][0];
#pragma unroll
        for (int i = 0; i < 4; ++i) {
            v16bf b = *(const v16bf*)&sB[buf][wn * 4 + i][lane][0];
            acc[i] = __builtin_amdgcn_wmma_f32_16x16x32_bf16(
                false, a, false, b, (short)0, acc[i], false, false);
        }

        if (kt + 1 < KT32) {
            wait_async0();
            __syncthreads();
        }
    }

    // C/D layout: VGPR r -> M = r + 8*(lane>=16), N = lane&15
    const int ncol = lane & 15;
    const int mhi  = (lane >> 4) ? 8 : 0;
#pragma unroll
    for (int i = 0; i < 4; ++i) {
        int ntflat = ntflat0 + wn * 4 + i;
        int gate   = ntflat / NT16;
        int nt     = ntflat % NT16;
        float* out = xg + (size_t)gate * BT * Nn;
#pragma unroll
        for (int r = 0; r < 8; ++r) {
            int row = m0 + wm * 16 + r + mhi;
            out[(size_t)row * Nn + nt * 16 + ncol] = acc[i][r];
        }
    }
}

// ---------------------------------------------------------------------------
// One LSTM time step:
//   pre[g] = xg[g][:,t,:] + h_{t-1} @ Wr[g]   (bf16 WMMA, f32 accum)
//   i,f,o = hard_sigmoid ; g = tanh ; c = c*f + i*g ; h = tanh(c)*o
// grid = (B/16, NT16), block = 128 (4 waves, wave == gate)
// ---------------------------------------------------------------------------
__global__ void lstm_step_kernel(const __bf16* __restrict__ hb_in,
                                 __bf16* __restrict__ hb_out,
                                 float* __restrict__ c,
                                 const float* __restrict__ xg,
                                 const __bf16* __restrict__ wrp,
                                 float* __restrict__ hs,
                                 int t) {
    const int lane = threadIdx.x & 31;
    const int gate = threadIdx.x >> 5;
    const int mt   = blockIdx.x;   // batch tile (16 rows)
    const int nt   = blockIdx.y;   // hidden-column tile (16 cols)

    __shared__ float gs[4][16][16];

    v8f acc;
#pragma unroll
    for (int r = 0; r < 8; ++r) acc[r] = 0.0f;

    const __bf16* bbase = wrp + (((size_t)gate * NT16 + nt) * KT32) * 512;

#pragma unroll 4
    for (int kt = 0; kt < KT32; ++kt) {
        v16bf a = load_a_frag(hb_in, mt * 16, Nn, kt * 32, lane);
        v16bf b = load_b_frag(bbase + (size_t)kt * 512, lane);
        acc = __builtin_amdgcn_wmma_f32_16x16x32_bf16(
            false, a, false, b, (short)0, acc, false, false);
    }

    const int ncol = lane & 15;
    const int mhi  = (lane >> 4) ? 8 : 0;
    const float* xgg = xg + (size_t)gate * BT * Nn;
#pragma unroll
    for (int r = 0; r < 8; ++r) {
        int m = r + mhi;
        int b = mt * 16 + m;
        float pre = acc[r] + xgg[((size_t)b * Tt + t) * Nn + nt * 16 + ncol];
        gs[gate][m][ncol] = pre;
    }
    __syncthreads();

    for (int cell = threadIdx.x; cell < 256; cell += 128) {
        int m = cell >> 4;
        int n = cell & 15;
        int brow = mt * 16 + m;
        int hcol = nt * 16 + n;
        float gi = hsig(gs[0][m][n]);
        float gf = hsig(gs[1][m][n]);
        float gg = tanhf(gs[2][m][n]);
        float go = hsig(gs[3][m][n]);
        size_t sidx = (size_t)brow * Nn + hcol;
        float cn = c[sidx] * gf + gi * gg;
        float hn = tanhf(cn) * go;
        c[sidx]      = cn;
        hb_out[sidx] = (__bf16)hn;
        hs[((size_t)brow * Tt + t) * Nn + hcol] = hn;
    }
}

// ---------------------------------------------------------------------------
// Host-side orchestration
// ---------------------------------------------------------------------------
extern "C" void kernel_launch(void* const* d_in, const int* in_sizes, int n_in,
                              void* d_out, int out_size, void* d_ws, size_t ws_size,
                              hipStream_t stream) {
    (void)in_sizes; (void)n_in; (void)out_size; (void)ws_size;

    const float* x  = (const float*)d_in[0];
    const float* h0 = (const float*)d_in[1];
    const float* c0 = (const float*)d_in[2];
    const float* Wk[4] = { (const float*)d_in[3], (const float*)d_in[5],
                           (const float*)d_in[7], (const float*)d_in[9] };
    const float* Wr[4] = { (const float*)d_in[4], (const float*)d_in[6],
                           (const float*)d_in[8], (const float*)d_in[10] };
    float* hs = (float*)d_out;

    uint8_t* ws = (uint8_t*)d_ws;
    __bf16* xb  = (__bf16*)(ws + OFF_XB);
    __bf16* wkp = (__bf16*)(ws + OFF_WKP);
    __bf16* wrp = (__bf16*)(ws + OFF_WRP);
    __bf16* hb  = (__bf16*)(ws + OFF_HB);     // double buffer [2][B][N]
    float*  cst = (float*)(ws + OFF_C);
    float*  xg  = (float*)(ws + OFF_XG);

    // 1) Convert x to bf16
    {
        int n = BT * Dd;
        cvt_bf16_kernel<<<(n + 255) / 256, 256, 0, stream>>>(x, xb, n);
    }

    // 2) Pack weights into WMMA B-fragment order (bf16)
    {
        const int total = NT16 * KT32 * 512;
        const int gridp = (total + 255) / 256;
        const size_t gstride = (size_t)Dd * Nn;
        for (int g = 0; g < 4; ++g) {
            pack_w_kernel<<<gridp, 256, 0, stream>>>(Wk[g], wkp + g * gstride);
            pack_w_kernel<<<gridp, 256, 0, stream>>>(Wr[g], wrp + g * gstride);
        }
    }

    // 3) Initialize h (bf16) and c (f32) state
    {
        int n = Bb * Nn;
        init_state_kernel<<<(n + 255) / 256, 256, 0, stream>>>(h0, c0, hb, cst);
    }

    // 4) Input projections for all 4 gates (flat-N WMMA GEMM, async staging)
    proj_gemm_kernel<<<dim3(BT / 64, NTFLAT / 8), 256, 0, stream>>>(xb, wkp, xg);

    // 5) Sequential recurrence: one kernel per time step (device-wide ordering
    //    via kernel boundaries; h double-buffered).
    const size_t hbuf = (size_t)Bb * Nn;
    for (int t = 0; t < Tt; ++t) {
        const __bf16* hin  = hb + (t & 1) * hbuf;
        __bf16*       hout = hb + ((t + 1) & 1) * hbuf;
        lstm_step_kernel<<<dim3(Bb / 16, NT16), 128, 0, stream>>>(
            hin, hout, cst, xg, wrp, hs, t);
    }
}
